// ind_lstm_55825984913699
// MI455X (gfx1250) — matrix operand order repaired and checked by reading, not compiled, not run on hardware
//
#include <hip/hip_runtime.h>
#include <hip/hip_bf16.h>

#define T_STEPS 16384
#define BLK 256          // input width / decoder hidden
#define SIG 128          // encoder hidden
#define GE  512          // 4*SIG encoder gate width
#define GD  1024         // 4*BLK decoder gate width
#define NSEG 64
#define SEGLEN 256       // NSEG*SEGLEN >= T_STEPS-1
#define LAMBDA 0.25f

typedef __attribute__((ext_vector_type(16))) __bf16 v16bf;
typedef __attribute__((ext_vector_type(8)))  float  v8f;

__device__ __forceinline__ int imin(int a, int b) { return a < b ? a : b; }
__device__ __forceinline__ float sigmoidf_(float x) { return 1.0f / (1.0f + __expf(-x)); }
__device__ __forceinline__ float tanh_fast(float x) { return 2.0f / (1.0f + __expf(-2.0f * x)) - 1.0f; }

// One 16x16x32 bf16 WMMA accumulation step.
// arow: this lane's A row (row index = lane&15), pre-offset to current K chunk.
// brow: this lane's weight row (output col n = lane&15), pre-offset to K chunk.
// A(m,k) = arow_m[k]; B(k,n) = brow_n[k]  (i.e. C = X * W^T).
__device__ __forceinline__ v8f wmma_k32(const float* __restrict__ arow,
                                        const float* __restrict__ brow,
                                        int half, v8f c) {
  v16bf a, b;
#pragma unroll
  for (int r = 0; r < 8; ++r) {
    // A 16-bit 16x32 layout: VGPR r holds K = (r/4)*16 + half*8 + (r%4)*2 + {0,1}
    const int ka = ((r >> 2) << 4) + (half << 3) + ((r & 3) << 1);
    float2 av = *(const float2*)(arow + ka);
    a[2 * r]     = (__bf16)av.x;
    a[2 * r + 1] = (__bf16)av.y;
    // B 16-bit 32x16 layout: VGPR r holds K = (r/4)*16 + half*8 + (r%4)*2 + {0,1}, N = lane&15
    const int kb = ((r >> 2) << 4) + (half << 3) + ((r & 3) << 1);
    float2 bv = *(const float2*)(brow + kb);
    b[2 * r]     = (__bf16)bv.x;
    b[2 * r + 1] = (__bf16)bv.y;
  }
  return __builtin_amdgcn_wmma_f32_16x16x32_bf16(false, a, false, b, (short)0, c,
                                                 false, false);
}

// xg_enc[t][n] = sum_k inputs[t][k]*We_ih[n][k] + (be_ih[n]+be_hh[n])
// grid (8, 1024), block 128 (4 waves, one 16x16 tile each)
__global__ void gemm_enc(const float* __restrict__ inputs,
                         const float* __restrict__ We_ih,
                         const float* __restrict__ be_ih,
                         const float* __restrict__ be_hh,
                         float* __restrict__ xg) {
  const int lane = threadIdx.x & 31;
  const int wave = threadIdx.x >> 5;
  const int half = lane >> 4, idx = lane & 15;
  const int n0 = (blockIdx.x * 4 + wave) * 16;
  const int t0 = blockIdx.y * 16;
  const float* arow = inputs + (size_t)(t0 + idx) * BLK;
  const float* brow = We_ih + (size_t)(n0 + idx) * BLK;
  v8f c = {};
#pragma unroll
  for (int k = 0; k < BLK; k += 32) c = wmma_k32(arow + k, brow + k, half, c);
  const float bias = be_ih[n0 + idx] + be_hh[n0 + idx];
#pragma unroll
  for (int r = 0; r < 8; ++r) {
    const int row = t0 + r + 8 * half;  // C layout: M = r + 8*half, N = idx
    xg[(size_t)row * GE + n0 + idx] = c[r] + bias;
  }
}

// Serial encoder LSTM. One workgroup; We_hh register-resident (128 VGPRs/lane).
__global__ void __launch_bounds__(512, 1)
encoder_scan(const float* __restrict__ xg, const float* __restrict__ We_hh,
             float* __restrict__ signal, float* __restrict__ accum) {
  __shared__ __align__(16) float h_lds[SIG];
  __shared__ float acc_lds[GE];
  const int j = threadIdx.x;
  float w[SIG];
#pragma unroll
  for (int k = 0; k < SIG; ++k) w[k] = We_hh[(size_t)j * SIG + k];
  float cst = 0.0f, l1 = 0.0f;
  if (j < SIG) h_lds[j] = 0.0f;
  __syncthreads();
  for (int t = 0; t < T_STEPS; ++t) {
    float acc = xg[(size_t)t * GE + j];
    const float4* h4 = (const float4*)h_lds;
#pragma unroll
    for (int k = 0; k < SIG / 4; ++k) {
      const float4 hv = h4[k];  // wave-uniform LDS broadcast
      acc += hv.x * w[4 * k] + hv.y * w[4 * k + 1] + hv.z * w[4 * k + 2] +
             hv.w * w[4 * k + 3];
    }
    acc_lds[j] = acc;
    __syncthreads();
    if (j < SIG) {
      const float ig = sigmoidf_(acc_lds[j]);
      const float fg = sigmoidf_(acc_lds[SIG + j]);
      const float gg = tanh_fast(acc_lds[2 * SIG + j]);
      const float og = sigmoidf_(acc_lds[3 * SIG + j]);
      cst = fg * cst + ig * gg;
      const float h = og * tanh_fast(cst);
      h_lds[j] = h;
      signal[(size_t)t * SIG + j] = h;
      l1 += fabsf(h);
    }
    __syncthreads();
  }
  if (j < SIG) acc_lds[j] = l1;
  __syncthreads();
  if (j == 0) {
    float s = 0.0f;
    for (int k = 0; k < SIG; ++k) s += acc_lds[k];
    accum[1] = s;  // sum |signal|
  }
}

// xg_dec[t][n] = signal[t+1]·Wd_ih[n] + inputs[t]·Wd_hh[n] + (bd_ih[n]+bd_hh[n])
// grid (16, 1024), block 128
__global__ void gemm_dec(const float* __restrict__ signal,
                         const float* __restrict__ inputs,
                         const float* __restrict__ Wd_ih,
                         const float* __restrict__ Wd_hh,
                         const float* __restrict__ bd_ih,
                         const float* __restrict__ bd_hh,
                         float* __restrict__ xg) {
  const int lane = threadIdx.x & 31;
  const int wave = threadIdx.x >> 5;
  const int half = lane >> 4, idx = lane & 15;
  const int n0 = (blockIdx.x * 4 + wave) * 16;
  const int t0 = blockIdx.y * 16;
  int tr = t0 + idx;
  if (tr > T_STEPS - 2) tr = T_STEPS - 2;  // clamp loads; stores guarded below
  const float* arow1 = signal + (size_t)(tr + 1) * SIG;
  const float* brow1 = Wd_ih + (size_t)(n0 + idx) * SIG;
  const float* arow2 = inputs + (size_t)tr * BLK;
  const float* brow2 = Wd_hh + (size_t)(n0 + idx) * BLK;
  v8f c = {};
#pragma unroll
  for (int k = 0; k < SIG; k += 32) c = wmma_k32(arow1 + k, brow1 + k, half, c);
#pragma unroll
  for (int k = 0; k < BLK; k += 32) c = wmma_k32(arow2 + k, brow2 + k, half, c);
  const float bias = bd_ih[n0 + idx] + bd_hh[n0 + idx];
#pragma unroll
  for (int r = 0; r < 8; ++r) {
    const int row = t0 + r + 8 * half;
    if (row < T_STEPS - 1) xg[(size_t)row * GD + n0 + idx] = c[r] + bias;
  }
}

// Decoder c-recurrence is linear given gates: c_t = f_t*c_{t-1} + i_t*g_t.
// Pass A: per-segment (P = prod f, B = affine offset).
__global__ void dec_passA(const float* __restrict__ xg, float* __restrict__ segP,
                          float* __restrict__ segB) {
  const int s = blockIdx.x, j = threadIdx.x;
  const int t0 = s * SEGLEN, t1 = imin(t0 + SEGLEN, T_STEPS - 1);
  float P = 1.0f, B = 0.0f;
  for (int t = t0; t < t1; ++t) {
    const float* g = xg + (size_t)t * GD;
    const float ig = sigmoidf_(g[j]);
    const float fg = sigmoidf_(g[BLK + j]);
    const float gg = tanh_fast(g[2 * BLK + j]);
    P *= fg;
    B = fg * B + ig * gg;
  }
  segP[s * BLK + j] = P;
  segB[s * BLK + j] = B;
}

// Pass B: tiny sequential prefix across segments (256 independent columns).
__global__ void dec_passB(const float* __restrict__ segP,
                          const float* __restrict__ segB,
                          float* __restrict__ cin) {
  const int j = threadIdx.x;
  float c = 0.0f;
  for (int s = 0; s < NSEG; ++s) {
    cin[s * BLK + j] = c;
    c = segP[s * BLK + j] * c + segB[s * BLK + j];
  }
}

// Pass C: replay with correct c_in, accumulate MSE vs inputs[t+1].
__global__ void dec_passC(const float* __restrict__ xg,
                          const float* __restrict__ inputs,
                          const float* __restrict__ cin,
                          float* __restrict__ accum) {
  __shared__ float red[BLK];
  const int s = blockIdx.x, j = threadIdx.x;
  const int t0 = s * SEGLEN, t1 = imin(t0 + SEGLEN, T_STEPS - 1);
  float c = cin[s * BLK + j], mse = 0.0f;
  for (int t = t0; t < t1; ++t) {
    const float* g = xg + (size_t)t * GD;
    const float ig = sigmoidf_(g[j]);
    const float fg = sigmoidf_(g[BLK + j]);
    const float gg = tanh_fast(g[2 * BLK + j]);
    const float og = sigmoidf_(g[3 * BLK + j]);
    c = fg * c + ig * gg;
    const float h = og * tanh_fast(c);
    const float d = h - inputs[(size_t)(t + 1) * BLK + j];
    mse += d * d;
  }
  red[j] = mse;
  __syncthreads();
  for (int off = BLK / 2; off > 0; off >>= 1) {
    if (j < off) red[j] += red[j + off];
    __syncthreads();
  }
  if (j == 0) atomicAdd(&accum[0], red[0]);
}

__global__ void zero_accum(float* __restrict__ accum) {
  if (threadIdx.x < 2) accum[threadIdx.x] = 0.0f;
}

__global__ void finalize_loss(const float* __restrict__ accum,
                              float* __restrict__ out) {
  const float mse = accum[0] / (float)((T_STEPS - 1) * BLK);
  const float l1 = accum[1] / (float)(T_STEPS * SIG);
  out[0] = mse + LAMBDA * l1;
  out[1] = mse;
  out[2] = l1;
}

extern "C" void kernel_launch(void* const* d_in, const int* in_sizes, int n_in,
                              void* d_out, int out_size, void* d_ws, size_t ws_size,
                              hipStream_t stream) {
  const float* inputs = (const float*)d_in[0];
  const float* We_ih = (const float*)d_in[1];
  const float* We_hh = (const float*)d_in[2];
  const float* be_ih = (const float*)d_in[3];
  const float* be_hh = (const float*)d_in[4];
  const float* Wd_ih = (const float*)d_in[5];
  const float* Wd_hh = (const float*)d_in[6];
  const float* bd_ih = (const float*)d_in[7];
  const float* bd_hh = (const float*)d_in[8];
  float* out = (float*)d_out;

  float* ws = (float*)d_ws;
  float* xg_enc = ws;                                   // T*512
  float* signal = xg_enc + (size_t)T_STEPS * GE;        // T*128
  float* xg_dec = signal + (size_t)T_STEPS * SIG;       // (T-1)*1024
  float* segP = xg_dec + (size_t)(T_STEPS - 1) * GD;    // NSEG*256
  float* segB = segP + NSEG * BLK;                      // NSEG*256
  float* cin = segB + NSEG * BLK;                       // NSEG*256
  float* accum = cin + NSEG * BLK;                      // 2

  zero_accum<<<1, 32, 0, stream>>>(accum);
  gemm_enc<<<dim3(8, 1024), 128, 0, stream>>>(inputs, We_ih, be_ih, be_hh, xg_enc);
  encoder_scan<<<1, 512, 0, stream>>>(xg_enc, We_hh, signal, accum);
  gemm_dec<<<dim3(16, 1024), 128, 0, stream>>>(signal, inputs, Wd_ih, Wd_hh,
                                               bd_ih, bd_hh, xg_dec);
  dec_passA<<<NSEG, BLK, 0, stream>>>(xg_dec, segP, segB);
  dec_passB<<<1, BLK, 0, stream>>>(segP, segB, cin);
  dec_passC<<<NSEG, BLK, 0, stream>>>(xg_dec, inputs, cin, accum);
  finalize_loss<<<1, 1, 0, stream>>>(accum, out);
}
